// DFlashAttentionAccum_10677288698213
// MI455X (gfx1250) — matrix operand order repaired and checked by reading, not compiled, not run on hardware
//
#include <hip/hip_runtime.h>

// ---------------- problem constants ----------------
#define HID   4096
#define Lq    128
#define CTXL  896
#define Tt    1024      // CTX + L
#define STATE 3072
#define Sa    4096      // STATE + T
#define HQ    32
#define HKV   8
#define Dh    128
#define NCH   4         // S-chunks for flash-decoding split
#define SCH   (Sa / NCH)
#define QPOS0 3968      // STATE + CTX
#define EPSN  1e-6f
#define SCALE_ATTN 0.08838834764831845f  // 128^-0.5

typedef __bf16 bf16_t;
typedef __attribute__((ext_vector_type(16))) __bf16 v16bf;
typedef __attribute__((ext_vector_type(8)))  float  v8f;

// ---------------- WMMA fragment loaders (ISA 7.12.2 layouts) ----------------
// A 16x32 bf16: lane L holds row m=L&15; k = (e&7) + (e<8?0:16) + 8*(L>=16)
__device__ inline v16bf frag_a_rowmajor(const bf16_t* __restrict__ p0, int ld, int lane) {
  int m  = lane & 15;
  int kh = (lane >> 4) << 3;           // 0 or 8
  const bf16_t* p = p0 + (size_t)m * ld + kh;
  v16bf f;
#pragma unroll
  for (int e = 0; e < 8; ++e) f[e] = p[e];
#pragma unroll
  for (int e = 0; e < 8; ++e) f[8 + e] = p[16 + e];
  return f;
}

// B 32x16 bf16, element (k,n) sourced from ROW-MAJOR src[n][k] (i.e. B = src^T):
// lane L holds col n=L&15; k = (e&7) + (e<8?0:16) + 8*(L>=16)  (matches A-frag).
__device__ inline v16bf frag_b_from_nk(const bf16_t* __restrict__ p0, int ld, int lane) {
  int n  = lane & 15;
  int kh = (lane >> 4) << 3;           // 0 or 8
  const bf16_t* p = p0 + (size_t)n * ld + kh;
  v16bf f;
#pragma unroll
  for (int e = 0; e < 8; ++e) f[e] = p[e];
#pragma unroll
  for (int e = 0; e < 8; ++e) f[8 + e] = p[16 + e];
  return f;
}

// ---------------- elementwise converts ----------------
__global__ void f2b_kernel(const float* __restrict__ s, bf16_t* __restrict__ d, int n) {
  int i = blockIdx.x * blockDim.x + threadIdx.x;
  if (i < n) d[i] = (bf16_t)s[i];
}

// K cache -> bf16 [HKV][S][D] (row-major in d; K stays key-major: QK^T B-frag reads rows)
__global__ void cacheK_bf16_kernel(const float* __restrict__ cK, bf16_t* __restrict__ Kall) {
  int i = blockIdx.x * blockDim.x + threadIdx.x;
  if (i >= HKV * STATE * Dh) return;
  int d = i & (Dh - 1);
  int s = (i / Dh) % STATE;
  int h = i / (Dh * STATE);
  Kall[((size_t)h * Sa + s) * Dh + d] = (bf16_t)cK[i];
}

// v_pre [t][h][d] -> f32 output (1,HKV,T,D)
__global__ void scatter_v_kernel(const float* __restrict__ vp, float* __restrict__ outv) {
  int i = blockIdx.x * blockDim.x + threadIdx.x;
  if (i >= Tt * HKV * Dh) return;
  int d = i & (Dh - 1);
  int h = (i / Dh) & (HKV - 1);
  int t = i / (Dh * HKV);
  outv[((size_t)h * Tt + t) * Dh + d] = vp[i];
}

// LDS-tiled transpose with f32->bf16: dst[h][c][r] = src[h][r][c]
// grid: (rows/32, cols/32, H); block 256
__global__ __launch_bounds__(256) void transpose_f2b_kernel(
    const float* __restrict__ src, size_t srcHstride, size_t src_ld,
    bf16_t* __restrict__ dst, size_t dstHstride, size_t dst_ld) {
  __shared__ bf16_t tile[32][33];
  const float* s = src + (size_t)blockIdx.z * srcHstride;
  bf16_t* d = dst + (size_t)blockIdx.z * dstHstride;
  int r0 = blockIdx.x * 32, c0 = blockIdx.y * 32;
  int tx = threadIdx.x & 31, ty = threadIdx.x >> 5;   // ty 0..7
#pragma unroll
  for (int k = 0; k < 4; ++k)
    tile[ty + 8 * k][tx] = (bf16_t)s[(size_t)(r0 + ty + 8 * k) * src_ld + c0 + tx];
  __syncthreads();
#pragma unroll
  for (int k = 0; k < 4; ++k)
    d[(size_t)(c0 + ty + 8 * k) * dst_ld + r0 + tx] = tile[tx][ty + 8 * k];
}

// ---------------- generic bf16 GEMM: C[M,N] = A[M,K] * B[N,K]^T ----------------
// one wave = 16x64 tile. Explicit ping-pong double buffering (unroll-2, peeled
// prologue/tail) so the register rotation costs zero v_mov copies and the next
// 64-deep K-slice's loads are in flight while the current WMMAs execute.
// Requires K % 64 == 0 and K >= 128 (all call sites use K = 4096).
__global__ __launch_bounds__(128) void gemm16x64_bf16(const bf16_t* __restrict__ A,
                                                      const bf16_t* __restrict__ B,
                                                      float* __restrict__ C,
                                                      int M, int N, int K) {
  int wave = (int)((blockIdx.x * blockDim.x + threadIdx.x) >> 5);
  int lane = threadIdx.x & 31;
  int ntn  = N >> 6;
  int tm = wave / ntn, tn = wave % ntn;
  if (tm * 16 >= M) return;
  const bf16_t* Ab = A + (size_t)(tm * 16) * K;
  const bf16_t* Bb = B + (size_t)(tn * 64) * K;
  v8f acc[4] = {{}, {}, {}, {}};

  v16bf a0, a1, b0[4], b1[4];
  a0 = frag_a_rowmajor(Ab, K, lane);
#pragma unroll
  for (int j = 0; j < 4; ++j) b0[j] = frag_b_from_nk(Bb + (size_t)j * 16 * K, K, lane);
  a1 = frag_a_rowmajor(Ab + 32, K, lane);
#pragma unroll
  for (int j = 0; j < 4; ++j) b1[j] = frag_b_from_nk(Bb + (size_t)j * 16 * K + 32, K, lane);

  for (int k0 = 64; k0 < K; k0 += 64) {
#pragma unroll
    for (int j = 0; j < 4; ++j)
      acc[j] = __builtin_amdgcn_wmma_f32_16x16x32_bf16(false, a0, false, b0[j],
                                                       (short)0, acc[j], false, false);
    a0 = frag_a_rowmajor(Ab + k0, K, lane);
#pragma unroll
    for (int j = 0; j < 4; ++j) b0[j] = frag_b_from_nk(Bb + (size_t)j * 16 * K + k0, K, lane);
#pragma unroll
    for (int j = 0; j < 4; ++j)
      acc[j] = __builtin_amdgcn_wmma_f32_16x16x32_bf16(false, a1, false, b1[j],
                                                       (short)0, acc[j], false, false);
    a1 = frag_a_rowmajor(Ab + k0 + 32, K, lane);
#pragma unroll
    for (int j = 0; j < 4; ++j) b1[j] = frag_b_from_nk(Bb + (size_t)j * 16 * K + k0 + 32, K, lane);
  }
#pragma unroll
  for (int j = 0; j < 4; ++j)
    acc[j] = __builtin_amdgcn_wmma_f32_16x16x32_bf16(false, a0, false, b0[j],
                                                     (short)0, acc[j], false, false);
#pragma unroll
  for (int j = 0; j < 4; ++j)
    acc[j] = __builtin_amdgcn_wmma_f32_16x16x32_bf16(false, a1, false, b1[j],
                                                     (short)0, acc[j], false, false);

  int half = lane >> 4, n = lane & 15;
#pragma unroll
  for (int j = 0; j < 4; ++j)
#pragma unroll
    for (int r = 0; r < 8; ++r)
      C[(size_t)(tm * 16 + r + 8 * half) * N + tn * 64 + j * 16 + n] = acc[j][r];
}

// ---------------- ane-norm + RoPE (per (row, head), D=128) ----------------
__device__ inline float block_sum128(float v, float* red) {
#pragma unroll
  for (int o = 16; o > 0; o >>= 1) v += __shfl_down(v, o, 32);
  int lane = threadIdx.x & 31, wv = threadIdx.x >> 5;
  if (lane == 0) red[wv] = v;
  __syncthreads();
  float s = red[0] + red[1] + red[2] + red[3];
  __syncthreads();
  return s;
}

__global__ __launch_bounds__(128) void norm_rope_kernel(
    const float* __restrict__ src, int NH,
    const float* __restrict__ w,
    const float* __restrict__ cosb, const float* __restrict__ sinb,
    float* __restrict__ out_f32, int rowsTot,          // may be null
    bf16_t* __restrict__ out_b16, int sRows, int sOff) {
  __shared__ float red[4];
  __shared__ float sh[Dh];
  int t = blockIdx.x / NH, h = blockIdx.x % NH, d = threadIdx.x;
  float x = src[((size_t)t * NH + h) * Dh + d];
  float mean = block_sum128(x, red) * (1.0f / Dh);
  float xc = x - mean;
  float var = block_sum128(xc * xc, red) * (1.0f / Dh);
  float y = xc * rsqrtf(var + EPSN) * w[d];
  sh[d] = y;
  __syncthreads();
  float rot = (d < Dh / 2) ? -sh[d + Dh / 2] : sh[d - Dh / 2];
  float res = y * cosb[(size_t)t * Dh + d] + rot * sinb[(size_t)t * Dh + d];
  if (out_f32) out_f32[((size_t)h * rowsTot + t) * Dh + d] = res;
  out_b16[((size_t)h * sRows + sOff + t) * Dh + d] = (bf16_t)res;
}

// ---------------- flash attention: one wave = (chunk, head, 16 q-rows) ----------------
// Emits per-chunk unnormalized O plus (m, l) partials; merged afterwards.
__global__ __launch_bounds__(128) void attn_kernel(const bf16_t* __restrict__ Q,     // [HQ][L][D]
                                                   const bf16_t* __restrict__ Kall,  // [HKV][S][D]
                                                   const bf16_t* __restrict__ Vt,    // [HKV][D][S]
                                                   float* __restrict__ Opart,        // [NCH][HQ][L][D]
                                                   float* __restrict__ Mpart,        // [NCH][HQ][L]
                                                   float* __restrict__ Lpart) {      // [NCH][HQ][L]
  __shared__ float Pls[4][16 * 32];
  int lane = threadIdx.x & 31, wv = threadIdx.x >> 5;
  int gw  = blockIdx.x * 4 + wv;           // 0..1023
  int c   = gw >> 8;                       // S-chunk 0..3
  int rem = gw & 255;
  int h   = rem >> 3, qt = rem & 7;
  int kvh = h >> 2;                        // REP = 4
  int half = lane >> 4, n = lane & 15;
  float* P = &Pls[wv][0];

  const bf16_t* Qp = Q + ((size_t)h * Lq + qt * 16) * Dh;
  v16bf qa[4];
#pragma unroll
  for (int cc = 0; cc < 4; ++cc) qa[cc] = frag_a_rowmajor(Qp + cc * 32, Dh, lane);

  const bf16_t* Kb = Kall + (size_t)kvh * Sa * Dh;   // [s][d]
  const bf16_t* Vb = Vt   + (size_t)kvh * Dh * Sa;   // [d][s]

  v8f o[8] = {{}, {}, {}, {}, {}, {}, {}, {}};
  float mrow[8], lrow[8];
#pragma unroll
  for (int r = 0; r < 8; ++r) { mrow[r] = -1e30f; lrow[r] = 0.0f; }

  int sBeg = c * SCH;
  for (int s0 = sBeg; s0 < sBeg + SCH; s0 += 32) {
    __builtin_prefetch((const void*)(Kb + (size_t)(s0 + 32) * Dh), 0, 1);
    __builtin_prefetch((const void*)(Vb + (size_t)(s0 + 32)), 0, 1);
    // ---- scores: 16 q-rows x 32 keys, K-dim = D in 4 chunks of 32 ----
    v8f sc[2] = {{}, {}};
#pragma unroll
    for (int sub = 0; sub < 2; ++sub) {
#pragma unroll
      for (int cc = 0; cc < 4; ++cc) {
        v16bf b = frag_b_from_nk(Kb + (size_t)(s0 + sub * 16) * Dh + cc * 32, Dh, lane);
        sc[sub] = __builtin_amdgcn_wmma_f32_16x16x32_bf16(false, qa[cc], false, b,
                                                          (short)0, sc[sub], false, false);
      }
    }
    // ---- scale + causal mask (key s valid iff s <= QPOS0 + l_global) ----
#pragma unroll
    for (int sub = 0; sub < 2; ++sub) {
#pragma unroll
      for (int r = 0; r < 8; ++r) {
        int lg = qt * 16 + r + 8 * half;
        int s  = s0 + sub * 16 + n;
        float v = sc[sub][r] * SCALE_ATTN;
        sc[sub][r] = (s <= QPOS0 + lg) ? v : -1e30f;
      }
    }
    // ---- online softmax update ----
#pragma unroll
    for (int r = 0; r < 8; ++r) {
      float m = fmaxf(sc[0][r], sc[1][r]);
#pragma unroll
      for (int xm = 8; xm > 0; xm >>= 1) m = fmaxf(m, __shfl_xor(m, xm, 32));
      float mn = fmaxf(mrow[r], m);
      float p0 = __expf(sc[0][r] - mn);
      float p1 = __expf(sc[1][r] - mn);
      sc[0][r] = p0; sc[1][r] = p1;
      float srow = p0 + p1;
#pragma unroll
      for (int xm = 8; xm > 0; xm >>= 1) srow += __shfl_xor(srow, xm, 32);
      float corr = __expf(mrow[r] - mn);
      lrow[r] = lrow[r] * corr + srow;
      mrow[r] = mn;
#pragma unroll
      for (int j = 0; j < 8; ++j) o[j][r] *= corr;
    }
    // ---- reshape P (C/D-frag) -> A-frag via LDS ----
    __syncthreads();
#pragma unroll
    for (int sub = 0; sub < 2; ++sub)
#pragma unroll
      for (int r = 0; r < 8; ++r)
        P[(r + 8 * half) * 32 + sub * 16 + n] = sc[sub][r];
    __syncthreads();
    v16bf pa;
    {
      int m  = lane & 15;
      int kh = half << 3;
#pragma unroll
      for (int e = 0; e < 8; ++e) pa[e]     = (bf16_t)P[m * 32 + kh + e];
#pragma unroll
      for (int e = 0; e < 8; ++e) pa[8 + e] = (bf16_t)P[m * 32 + kh + 16 + e];
    }
    // ---- O += P * V ; V^T layout makes B-frags contiguous per lane ----
#pragma unroll
    for (int j = 0; j < 8; ++j) {
      v16bf b = frag_b_from_nk(Vb + (size_t)(j * 16) * Sa + s0, Sa, lane);
      o[j] = __builtin_amdgcn_wmma_f32_16x16x32_bf16(false, pa, false, b,
                                                     (short)0, o[j], false, false);
    }
  }
  // ---- epilogue: write unnormalized partials ----
  float* Obase = Opart + ((size_t)(c * HQ + h) * Lq + qt * 16) * Dh;
#pragma unroll
  for (int r = 0; r < 8; ++r) {
#pragma unroll
    for (int j = 0; j < 8; ++j)
      Obase[(size_t)(r + 8 * half) * Dh + j * 16 + n] = o[j][r];
  }
  if (n == 0) {
#pragma unroll
    for (int r = 0; r < 8; ++r) {
      int row = (c * HQ + h) * Lq + qt * 16 + r + 8 * half;
      Mpart[row] = mrow[r];
      Lpart[row] = lrow[r];
    }
  }
}

// ---------------- merge S-chunk partials -> o_bf16 [L][HQ*D] ----------------
__global__ void attn_merge_kernel(const float* __restrict__ Op,
                                  const float* __restrict__ Mp,
                                  const float* __restrict__ Lp,
                                  bf16_t* __restrict__ Ob) {
  int i = blockIdx.x * blockDim.x + threadIdx.x;   // over HQ*L*D
  if (i >= HQ * Lq * Dh) return;
  int d = i & (Dh - 1);
  int l = (i / Dh) & (Lq - 1);
  int h = i / (Dh * Lq);
  int row = h * Lq + l;
  float m[NCH], w[NCH];
  float M = -1e30f;
#pragma unroll
  for (int cc = 0; cc < NCH; ++cc) { m[cc] = Mp[cc * HQ * Lq + row]; M = fmaxf(M, m[cc]); }
  float Ltot = 0.0f, num = 0.0f;
#pragma unroll
  for (int cc = 0; cc < NCH; ++cc) {
    w[cc] = __expf(m[cc] - M);
    Ltot += Lp[cc * HQ * Lq + row] * w[cc];
    num  += Op[((size_t)cc * HQ * Lq + row) * Dh + d] * w[cc];
  }
  Ob[(size_t)l * (HQ * Dh) + h * Dh + d] = (bf16_t)(num / Ltot);
}

// ---------------- host orchestration ----------------
extern "C" void kernel_launch(void* const* d_in, const int* in_sizes, int n_in,
                              void* d_out, int out_size, void* d_ws, size_t ws_size,
                              hipStream_t stream) {
  (void)in_sizes; (void)n_in; (void)out_size; (void)ws_size;
  const float* x      = (const float*)d_in[0];
  const float* x_ctx  = (const float*)d_in[1];
  const float* cos_q  = (const float*)d_in[2];
  const float* sin_q  = (const float*)d_in[3];
  const float* cos_k  = (const float*)d_in[4];
  const float* sin_k  = (const float*)d_in[5];
  const float* cacheK = (const float*)d_in[6];
  const float* cacheV = (const float*)d_in[7];
  // d_in[8] = causal_mask (bool) -- handled analytically
  const float* Wq = (const float*)d_in[9];
  const float* Wk = (const float*)d_in[10];
  const float* Wv = (const float*)d_in[11];
  const float* Wo = (const float*)d_in[12];
  const float* qw = (const float*)d_in[13];
  const float* kw = (const float*)d_in[14];

  float* out_x = (float*)d_out;                       // (1,L,HID)
  float* out_k = out_x + (size_t)Lq * HID;            // (1,HKV,T,D)
  float* out_v = out_k + (size_t)HKV * Tt * Dh;       // (1,HKV,T,D)

  char* p = (char*)d_ws;
  auto carve = [&](size_t bytes) {
    void* r = (void*)p;
    p += (bytes + 255) & ~(size_t)255;
    return r;
  };
  bf16_t* c_b   = (bf16_t*)carve((size_t)Tt * HID * 2);
  bf16_t* Wq_b  = (bf16_t*)carve((size_t)HQ * Dh * HID * 2);
  bf16_t* Wk_b  = (bf16_t*)carve((size_t)HKV * Dh * HID * 2);
  bf16_t* Wv_b  = (bf16_t*)carve((size_t)HKV * Dh * HID * 2);
  bf16_t* Wo_b  = (bf16_t*)carve((size_t)HID * HQ * Dh * 2);
  float*  q_pre = (float*)carve((size_t)Lq * HQ * Dh * 4);
  float*  k_pre = (float*)carve((size_t)Tt * HKV * Dh * 4);
  float*  v_pre = (float*)carve((size_t)Tt * HKV * Dh * 4);
  bf16_t* Qb    = (bf16_t*)carve((size_t)HQ * Lq * Dh * 2);
  bf16_t* Kall  = (bf16_t*)carve((size_t)HKV * Sa * Dh * 2);
  bf16_t* Vt    = (bf16_t*)carve((size_t)HKV * Dh * Sa * 2);   // V transposed [h][d][s]
  bf16_t* Ob    = (bf16_t*)carve((size_t)Lq * HQ * Dh * 2);
  float*  Opart = (float*)carve((size_t)NCH * HQ * Lq * Dh * 4);
  float*  Mpart = (float*)carve((size_t)NCH * HQ * Lq * 4);
  float*  Lpart = (float*)carve((size_t)NCH * HQ * Lq * 4);

  const int THR = 256;
  auto cgrid = [](int n) { return (n + 255) / 256; };

  // 1) bf16 conversions (weights then live in the 192MB L2)
  f2b_kernel<<<cgrid(CTXL * HID), THR, 0, stream>>>(x_ctx, c_b, CTXL * HID);
  f2b_kernel<<<cgrid(Lq * HID), THR, 0, stream>>>(x, c_b + (size_t)CTXL * HID, Lq * HID);
  f2b_kernel<<<cgrid(HQ * Dh * HID), THR, 0, stream>>>(Wq, Wq_b, HQ * Dh * HID);
  f2b_kernel<<<cgrid(HKV * Dh * HID), THR, 0, stream>>>(Wk, Wk_b, HKV * Dh * HID);
  f2b_kernel<<<cgrid(HKV * Dh * HID), THR, 0, stream>>>(Wv, Wv_b, HKV * Dh * HID);
  f2b_kernel<<<cgrid(HID * HQ * Dh), THR, 0, stream>>>(Wo, Wo_b, HID * HQ * Dh);
  cacheK_bf16_kernel<<<cgrid(HKV * STATE * Dh), THR, 0, stream>>>(cacheK, Kall);
  // V cache -> Vt[h][d][s] (s < STATE), LDS-tiled transpose
  transpose_f2b_kernel<<<dim3(STATE / 32, Dh / 32, HKV), 256, 0, stream>>>(
      cacheV, (size_t)STATE * Dh, Dh, Vt, (size_t)Dh * Sa, Sa);

  // 2) projections (WMMA bf16, f32 accumulate, ping-pong pipelined)
  {
    int waves = (Lq / 16) * ((HQ * Dh) / 64);   // 512
    gemm16x64_bf16<<<(waves + 3) / 4, 128, 0, stream>>>(
        c_b + (size_t)CTXL * HID, Wq_b, q_pre, Lq, HQ * Dh, HID);
  }
  {
    int waves = (Tt / 16) * ((HKV * Dh) / 64);  // 1024
    gemm16x64_bf16<<<(waves + 3) / 4, 128, 0, stream>>>(c_b, Wk_b, k_pre, Tt, HKV * Dh, HID);
    gemm16x64_bf16<<<(waves + 3) / 4, 128, 0, stream>>>(c_b, Wv_b, v_pre, Tt, HKV * Dh, HID);
  }

  // 3) ane-norm + RoPE; f32 outputs + bf16 attention operands
  norm_rope_kernel<<<Lq * HQ, 128, 0, stream>>>(q_pre, HQ, qw, cos_q, sin_q,
                                                nullptr, 0, Qb, Lq, 0);
  norm_rope_kernel<<<Tt * HKV, 128, 0, stream>>>(k_pre, HKV, kw, cos_k, sin_k,
                                                 out_k, Tt, Kall, Sa, STATE);
  scatter_v_kernel<<<cgrid(Tt * HKV * Dh), THR, 0, stream>>>(v_pre, out_v);
  // v_pre [t][h*D+d] -> Vt[h][d][STATE + t]
  transpose_f2b_kernel<<<dim3(Tt / 32, Dh / 32, HKV), 256, 0, stream>>>(
      v_pre, (size_t)Dh, (size_t)HKV * Dh, Vt + STATE, (size_t)Dh * Sa, Sa);

  // 4) flash attention: 4 S-chunks x 32 heads x 8 q-tiles = 1024 waves
  attn_kernel<<<256, 128, 0, stream>>>(Qb, Kall, Vt, Opart, Mpart, Lpart);
  attn_merge_kernel<<<cgrid(HQ * Lq * Dh), THR, 0, stream>>>(Opart, Mpart, Lpart, Ob);

  // 5) output projection
  {
    int waves = (Lq / 16) * (HID / 64);         // 512
    gemm16x64_bf16<<<(waves + 3) / 4, 128, 0, stream>>>(Ob, Wo_b, out_x, Lq, HID, HQ * Dh);
  }
}
